// SelfAttention_v2_83717502534318
// MI455X (gfx1250) — matrix operand order repaired and checked by reading, not compiled, not run on hardware
//
#include <hip/hip_runtime.h>
#include <stdint.h>

#define SEQ 4096
#define DIM 1024   // d_in == d_out

typedef __attribute__((ext_vector_type(16))) __bf16 v16bf;
typedef __attribute__((ext_vector_type(8)))  __bf16 v8bf;
typedef __attribute__((ext_vector_type(8)))  float  v8f;
typedef int v4i __attribute__((vector_size(16)));

union AFrag { v16bf v; v8bf h[2]; };

__device__ __forceinline__ __bf16 f2bf(float f) {
  union { float f; uint32_t u; } a; a.f = f;
  uint32_t r = a.u + 0x7FFFu + ((a.u >> 16) & 1u);   // round-to-nearest-even
  uint16_t h = (uint16_t)(r >> 16);
  __bf16 o;
  __builtin_memcpy(&o, &h, 2);
  return o;
}

__device__ __forceinline__ v8f wmma_bf16(v16bf a, v16bf b, v8f c) {
  // (neg_a, A, neg_b, B, c_mod, C, reuse_a, reuse_b)
  return __builtin_amdgcn_wmma_f32_16x16x32_bf16(false, a, false, b, (short)0, c, false, false);
}

// ---------------------------------------------------------------- fp32 -> bf16
__global__ void cvt_bf16_kernel(const float* __restrict__ src,
                                __bf16* __restrict__ dst, int n) {
  int i = blockIdx.x * blockDim.x + threadIdx.x;
  int stride = gridDim.x * blockDim.x;
  for (; i < n; i += stride) dst[i] = f2bf(src[i]);
}

// ------------------------------------------------- projections: C = Xb @ W^T
// W layout [d_out, d_in]: C[s,o] = sum_k Xb[s,k] * W[o,k]  (both row-major in k)
// grid (DIM/64, SEQ/128, 3), block 256 (8 waves). Wave -> 16x64 output strip.
// z==0 -> Qb[s,o]; z==1 -> Kb[s,o]; z==2 -> Vt[o,s] (transposed for P@V stage)
__global__ void proj_kernel(const __bf16* __restrict__ Xb,
                            const __bf16* __restrict__ Wq,
                            const __bf16* __restrict__ Wk,
                            const __bf16* __restrict__ Wv,
                            __bf16* __restrict__ Qb,
                            __bf16* __restrict__ Kb,
                            __bf16* __restrict__ Vt) {
  const int lane = threadIdx.x & 31;
  const int wave = threadIdx.x >> 5;
  const int m0 = blockIdx.y * 128 + wave * 16;
  const int n0 = blockIdx.x * 64;
  const int z  = blockIdx.z;
  const __bf16* __restrict__ W = (z == 0) ? Wq : (z == 1) ? Wk : Wv;
  __bf16* __restrict__ dst = (z == 0) ? Qb : Kb;

  v8f acc[4];
#pragma unroll
  for (int t = 0; t < 4; ++t) acc[t] = (v8f)(0.0f);

  for (int k0 = 0; k0 < DIM; k0 += 32) {
    // A fragment: lane = row m, one contiguous 16-elem K strip (k, k+8)
    AFrag ua;
    const __bf16* ap = Xb + (size_t)(m0 + (lane & 15)) * DIM + k0 + ((lane >> 4) << 4);
    ua.h[0] = *(const v8bf*)ap;
    ua.h[1] = *(const v8bf*)(ap + 8);
#pragma unroll
    for (int t = 0; t < 4; ++t) {
      // B fragment: lane = col n, one contiguous 16-elem K strip
      const __bf16* bp = W + (size_t)(n0 + t * 16 + (lane & 15)) * DIM
                           + k0 + ((lane >> 4) << 4);
      v16bf b = *(const v16bf*)bp;
      acc[t] = wmma_bf16(ua.v, b, acc[t]);
    }
  }

#pragma unroll
  for (int t = 0; t < 4; ++t) {
#pragma unroll
    for (int r = 0; r < 8; ++r) {
      int row = m0 + ((lane >> 4) << 3) + r;  // C/D layout: VGPR r -> M, lane -> N
      int col = n0 + t * 16 + (lane & 15);
      __bf16 val = f2bf(acc[t][r]);
      if (z == 2) Vt[(size_t)col * SEQ + row] = val;
      else        dst[(size_t)row * DIM + col] = val;
    }
  }
}

// --------------------------------------------- fused flash attention forward
// One workgroup (8 waves) owns 16 query rows. Iterate over keys in tiles of
// 128 (wave w computes the 16x16 score tile for keys [jt*128 + w*16, +16)).
// Online softmax stats in LDS, reductions parallelized across all 256
// threads; wave w accumulates context columns [w*128, w*128+128) in 8 v8f
// accumulators.
__global__ void attn_kernel(const __bf16* __restrict__ Qb,
                            const __bf16* __restrict__ Kb,
                            const __bf16* __restrict__ Vt,
                            float* __restrict__ out) {
  __shared__ __bf16 Qs[16 * DIM];     // 32 KB query tile
  __shared__ float  Ss[16 * 128];     // 8 KB scaled scores
  __shared__ __bf16 Ps[16 * 128];     // 4 KB softmax probabilities
  __shared__ float  red[16 * 16];     // 1 KB partial max/sum reduction
  __shared__ float  row_m[16], row_l[16], row_alpha[16];

  const int tid  = threadIdx.x;
  const int lane = tid & 31;
  const int wave = tid >> 5;
  const int q0 = blockIdx.x * 16;
  const float scale = 0.03125f;       // 1/sqrt(1024)

  // stage Q tile into LDS (32 KB) via CDNA5 async memory->LDS path
  {
    const __bf16* gsrc = Qb + (size_t)q0 * DIM;
#if __has_builtin(__builtin_amdgcn_global_load_async_to_lds_b128)
    typedef __attribute__((address_space(1))) v4i gv4i;
    typedef __attribute__((address_space(3))) v4i lv4i;
    gv4i* g = (gv4i*)gsrc;
    lv4i* l = (lv4i*)(&Qs[0]);
#pragma unroll
    for (int i = 0; i < 8; ++i) {     // 2048 x 16B, 8 per thread
      int idx = tid + i * 256;
      __builtin_amdgcn_global_load_async_to_lds_b128(g + idx, l + idx, 0, 0);
    }
#if __has_builtin(__builtin_amdgcn_s_wait_asynccnt)
    __builtin_amdgcn_s_wait_asynccnt(0);
#else
    asm volatile("s_wait_asynccnt 0" ::: "memory");
#endif
#else
    const uint4* __restrict__ s4 = (const uint4*)gsrc;
    uint4* d4 = (uint4*)Qs;
    for (int i = tid; i < 2048; i += 256) d4[i] = s4[i];
#endif
  }
  if (tid < 16) { row_m[tid] = -1e30f; row_l[tid] = 0.0f; }
  __syncthreads();

  v8f acc[8];
#pragma unroll
  for (int t = 0; t < 8; ++t) acc[t] = (v8f)(0.0f);

  // softmax work partition: thread -> (srow, 8 columns)
  const int srow = tid >> 4;          // 0..15
  const int schunk = tid & 15;        // 0..15 -> cols [schunk*8, +8)

  for (int jt = 0; jt < SEQ / 128; ++jt) {
    const int jbase = jt * 128;

    // prefetch next key tile rows into the cache hierarchy
    if (jt + 1 < SEQ / 128) {
      const __bf16* pf = Kb + (size_t)(jbase + 128 + wave * 16 + (lane & 15)) * DIM;
      __builtin_prefetch(pf, 0, 0);
    }

    // ---- phase 1: scores S = Q @ K^T for this wave's 16 keys (full depth)
    v8f sacc = (v8f)(0.0f);
    for (int k0 = 0; k0 < DIM; k0 += 32) {
      AFrag ua;
      const __bf16* ap = Qs + (lane & 15) * DIM + k0 + ((lane >> 4) << 4);
      ua.h[0] = *(const v8bf*)ap;
      ua.h[1] = *(const v8bf*)(ap + 8);
      const __bf16* bp = Kb + (size_t)(jbase + wave * 16 + (lane & 15)) * DIM
                            + k0 + ((lane >> 4) << 4);
      v16bf b = *(const v16bf*)bp;
      sacc = wmma_bf16(ua.v, b, sacc);
    }
#pragma unroll
    for (int r = 0; r < 8; ++r) {
      int row = ((lane >> 4) << 3) + r;
      int col = wave * 16 + (lane & 15);
      Ss[row * 128 + col] = sacc[r] * scale;
    }
    __syncthreads();

    // ---- phase 2a: per-thread partial max over 8 columns
    {
      float pmax = -1e30f;
#pragma unroll
      for (int i = 0; i < 8; ++i)
        pmax = fmaxf(pmax, Ss[srow * 128 + schunk * 8 + i]);
      red[tid] = pmax;
    }
    __syncthreads();

    // ---- phase 2b: row max / alpha (16 threads, 16-way reduce each)
    if (tid < 16) {
      float m_old = row_m[tid];
      float mx = m_old;
#pragma unroll
      for (int j = 0; j < 16; ++j) mx = fmaxf(mx, red[tid * 16 + j]);
      row_alpha[tid] = __expf(m_old - mx);
      row_m[tid] = mx;
    }
    __syncthreads();

    // ---- phase 2c: probabilities + partial sums (all 256 threads)
    {
      float mx = row_m[srow];
      float psum = 0.0f;
#pragma unroll
      for (int i = 0; i < 8; ++i) {
        float p = __expf(Ss[srow * 128 + schunk * 8 + i] - mx);
        Ps[srow * 128 + schunk * 8 + i] = f2bf(p);
        psum += p;
      }
      red[tid] = psum;
    }
    __syncthreads();

    // ---- phase 2d: running denominator (16 threads)
    if (tid < 16) {
      float l = row_l[tid] * row_alpha[tid];
#pragma unroll
      for (int j = 0; j < 16; ++j) l += red[tid * 16 + j];
      row_l[tid] = l;
    }
    // no barrier needed here: phase 3 reads only Ps (sync'd) and row_alpha
    // (sync'd); row_l is re-read only after the loop-end barrier.

    // ---- phase 3: rescale context, accumulate P @ V for this wave's columns
    float al[8];
#pragma unroll
    for (int r = 0; r < 8; ++r) al[r] = row_alpha[((lane >> 4) << 3) + r];
#pragma unroll
    for (int t = 0; t < 8; ++t)
#pragma unroll
      for (int r = 0; r < 8; ++r) acc[t][r] *= al[r];

#pragma unroll
    for (int ks = 0; ks < 4; ++ks) {            // 128 keys in 4 chunks of 32
      AFrag ua;
      const __bf16* ap = Ps + (lane & 15) * 128 + ks * 32 + ((lane >> 4) << 4);
      ua.h[0] = *(const v8bf*)ap;
      ua.h[1] = *(const v8bf*)(ap + 8);
#pragma unroll
      for (int t = 0; t < 8; ++t) {
        int ocol = wave * 128 + t * 16 + (lane & 15);
        const __bf16* bp = Vt + (size_t)ocol * SEQ + jbase + ks * 32
                              + ((lane >> 4) << 4);
        v16bf b = *(const v16bf*)bp;
        acc[t] = wmma_bf16(ua.v, b, acc[t]);
      }
    }
    __syncthreads();   // protect Ss/Ps/red/row_* before next iteration
  }

  // ---- finalize: divide by softmax denominator, write fp32 context
  float linv[8];
#pragma unroll
  for (int r = 0; r < 8; ++r) linv[r] = 1.0f / row_l[((lane >> 4) << 3) + r];
#pragma unroll
  for (int t = 0; t < 8; ++t)
#pragma unroll
    for (int r = 0; r < 8; ++r) {
      int row = q0 + ((lane >> 4) << 3) + r;
      int col = wave * 128 + t * 16 + (lane & 15);
      out[(size_t)row * DIM + col] = acc[t][r] * linv[r];
    }
}

// ---------------------------------------------------------------------------
extern "C" void kernel_launch(void* const* d_in, const int* in_sizes, int n_in,
                              void* d_out, int out_size, void* d_ws, size_t ws_size,
                              hipStream_t stream) {
  const float* x  = (const float*)d_in[0];
  const float* wq = (const float*)d_in[1];
  const float* wk = (const float*)d_in[2];
  const float* wv = (const float*)d_in[3];
  float* out = (float*)d_out;

  // workspace layout (bf16), total 38 MB
  __bf16* xb  = (__bf16*)d_ws;
  __bf16* wqb = xb  + (size_t)SEQ * DIM;
  __bf16* wkb = wqb + (size_t)DIM * DIM;
  __bf16* wvb = wkb + (size_t)DIM * DIM;
  __bf16* Qb  = wvb + (size_t)DIM * DIM;
  __bf16* Kb  = Qb  + (size_t)SEQ * DIM;
  __bf16* Vt  = Kb  + (size_t)SEQ * DIM;

  cvt_bf16_kernel<<<512, 256, 0, stream>>>(x,  xb,  SEQ * DIM);
  cvt_bf16_kernel<<<256, 256, 0, stream>>>(wq, wqb, DIM * DIM);
  cvt_bf16_kernel<<<256, 256, 0, stream>>>(wk, wkb, DIM * DIM);
  cvt_bf16_kernel<<<256, 256, 0, stream>>>(wv, wvb, DIM * DIM);

  proj_kernel<<<dim3(DIM / 64, SEQ / 128, 3), 256, 0, stream>>>(
      xb, wqb, wkb, wvb, Qb, Kb, Vt);

  attn_kernel<<<SEQ / 16, 256, 0, stream>>>(Qb, Kb, Vt, out);
}
